// Attn_56092272886097
// MI455X (gfx1250) — matrix operand order repaired and checked by reading, not compiled, not run on hardware
//
#include <hip/hip_runtime.h>
#include <hip/hip_bf16.h>

#define BATCH   2
#define SEQ     2048
#define DMODEL  2048
#define N_HEADS 16
#define DHEAD   128

typedef __bf16 bf16_t;
typedef __attribute__((ext_vector_type(16))) __bf16 v16bf;
typedef __attribute__((ext_vector_type(8)))  __bf16 v8bf;
typedef __attribute__((ext_vector_type(8)))  float  v8f;
typedef __attribute__((ext_vector_type(4)))  unsigned int u32x4;
typedef __attribute__((ext_vector_type(8)))  int i32x8;
typedef __attribute__((ext_vector_type(4)))  int i32x4;

#if __has_builtin(__builtin_amdgcn_tensor_load_to_lds)
#define HAVE_TDM 1
#else
#define HAVE_TDM 0
#endif

static __device__ __forceinline__ v8f wmma_bf16(v16bf a, v16bf b, v8f c) {
    return __builtin_amdgcn_wmma_f32_16x16x32_bf16(
        false, a, false, b, (short)0, c, false, false);
}

static __device__ __forceinline__ v16bf cat8(v8bf lo, v8bf hi) {
    return __builtin_shufflevector(lo, hi, 0,1,2,3,4,5,6,7,8,9,10,11,12,13,14,15);
}

#if HAVE_TDM
// byte offset of a __shared__ object within the workgroup LDS allocation
static __device__ __forceinline__ unsigned lds_addr_of(const void* p) {
    return (unsigned)(unsigned long long)(__attribute__((address_space(3))) const void*)p;
}

// 2-D TDM load: tile (tile_d1 rows x tile_d0 elems) of 2-byte elements,
// row stride stride0 elems, packed contiguously into LDS at lds_byte.
static __device__ __forceinline__ void tdm_load_2d_b16(
        unsigned lds_byte, unsigned long long gaddr,
        unsigned tensor_d0, unsigned tensor_d1,
        unsigned tile_d0, unsigned tile_d1, unsigned stride0) {
    u32x4 g0;
    g0.x = 0x1u;                                        // count=1 (valid), user mode
    g0.y = lds_byte;                                    // lds_addr
    g0.z = (unsigned)(gaddr & 0xFFFFFFFFull);           // global_addr[31:0]
    g0.w = (unsigned)((gaddr >> 32) & 0x1FFFFFFull)     // global_addr[56:32]
         | (2u << 30);                                  // type=2 ("image")
    i32x8 g1;
    g1[0] = (int)(1u << 16);                            // wg_mask=0, data_size=1 (2B)
    g1[1] = (int)((tensor_d0 & 0xFFFFu) << 16);         // tensor_dim0[15:0] @ bit48
    g1[2] = (int)(((tensor_d0 >> 16) & 0xFFFFu)         // tensor_dim0[31:16]
         | ((tensor_d1 & 0xFFFFu) << 16));              // tensor_dim1[15:0] @ bit80
    g1[3] = (int)(((tensor_d1 >> 16) & 0xFFFFu)         // tensor_dim1[31:16]
         | ((tile_d0 & 0xFFFFu) << 16));                // tile_dim0 @ bit112
    g1[4] = (int)(tile_d1 & 0xFFFFu);                   // tile_dim1, tile_dim2=0
    g1[5] = (int)stride0;                               // tensor_dim0_stride[31:0]
    g1[6] = 0;                                          // stride hi / dim1_stride lo
    g1[7] = 0;                                          // dim1_stride hi
    i32x4 z4 = {0, 0, 0, 0};
#if defined(__clang_major__) && __clang_major__ >= 23
    i32x8 z8 = {0, 0, 0, 0, 0, 0, 0, 0};
    __builtin_amdgcn_tensor_load_to_lds(g0, g1, z4, z4, z8, 0);
#else
    __builtin_amdgcn_tensor_load_to_lds(g0, g1, z4, z4, 0);
#endif
}

static __device__ __forceinline__ void wait_tensorcnt0() {
#if __has_builtin(__builtin_amdgcn_s_wait_tensorcnt)
    __builtin_amdgcn_s_wait_tensorcnt(0);
#else
    asm volatile("s_wait_tensorcnt 0x0" ::: "memory");
#endif
}
#endif  // HAVE_TDM

// ---------------------------------------------------------------- cvt f32->bf16
__global__ void cvt_f32_bf16(const float* __restrict__ in, bf16_t* __restrict__ out, int n) {
    int i = blockIdx.x * blockDim.x + threadIdx.x;
    int stride = gridDim.x * blockDim.x;
    for (; i < n; i += stride) out[i] = (bf16_t)in[i];
}

// ---------------------------------------------------------------- GEMM  C[M,N] = A[M,K] * Bw[N,K]^T
// block tile 128x128, BK=64, double-buffered LDS; wave 0 drives the Tensor Data
// Mover for the next K-slice while all 8 waves run WMMAs on the current slice.
__launch_bounds__(256)
__global__ void gemm_bf16_tn(const bf16_t* __restrict__ A,
                             const bf16_t* __restrict__ Bw,
                             float* __restrict__ C,
                             int M, int N, int K) {
    __shared__ __align__(32) bf16_t As[2][128][64];
    __shared__ __align__(32) bf16_t Bs[2][128][64];

    const int tid  = threadIdx.x;
    const int lane = tid & 31;
    const int wave = tid >> 5;
    const int wrow = wave & 3;
    const int wcol = wave >> 2;
    const int hl   = lane >> 4;
    const int l16  = lane & 15;
    const int m0   = blockIdx.y * 128;
    const int n0   = blockIdx.x * 128;

    v8f acc[2][4];
    {
        v8f z = {};
        #pragma unroll
        for (int mi = 0; mi < 2; ++mi)
            #pragma unroll
            for (int ni = 0; ni < 4; ++ni) acc[mi][ni] = z;
    }

#if HAVE_TDM
    unsigned asl[2] = { lds_addr_of(&As[0][0][0]), lds_addr_of(&As[1][0][0]) };
    unsigned bsl[2] = { lds_addr_of(&Bs[0][0][0]), lds_addr_of(&Bs[1][0][0]) };
    // prologue: DMA first K-slice into buffer 0
    if (wave == 0) {
        tdm_load_2d_b16(asl[0], (unsigned long long)(size_t)&A [(size_t)m0 * K],
                        64, 128, 64, 128, (unsigned)K);
        tdm_load_2d_b16(bsl[0], (unsigned long long)(size_t)&Bw[(size_t)n0 * K],
                        64, 128, 64, 128, (unsigned)K);
        wait_tensorcnt0();
    }
    __syncthreads();
#else
    {
        v8bf ta[4], tb[4];
        #pragma unroll
        for (int i = 0; i < 4; ++i) {
            int idx = i * 256 + tid;
            int row = idx >> 3, c8 = idx & 7;
            ta[i] = *(const v8bf*)&A [(size_t)(m0 + row) * K + c8 * 8];
            tb[i] = *(const v8bf*)&Bw[(size_t)(n0 + row) * K + c8 * 8];
        }
        #pragma unroll
        for (int i = 0; i < 4; ++i) {
            int idx = i * 256 + tid;
            int row = idx >> 3, c8 = idx & 7;
            *(v8bf*)&As[0][row][c8 * 8] = ta[i];
            *(v8bf*)&Bs[0][row][c8 * 8] = tb[i];
        }
        __syncthreads();
    }
#endif

    int buf = 0;
    for (int k0 = 0; k0 < K; k0 += 64) {
        const bool haveNext = (k0 + 64) < K;
#if HAVE_TDM
        // kick off DMA of the next slice into the other buffer
        if (haveNext && wave == 0) {
            tdm_load_2d_b16(asl[buf ^ 1],
                            (unsigned long long)(size_t)&A [(size_t)m0 * K + k0 + 64],
                            64, 128, 64, 128, (unsigned)K);
            tdm_load_2d_b16(bsl[buf ^ 1],
                            (unsigned long long)(size_t)&Bw[(size_t)n0 * K + k0 + 64],
                            64, 128, 64, 128, (unsigned)K);
        }
#else
        v8bf ta[4], tb[4];
        if (haveNext) {
            #pragma unroll
            for (int i = 0; i < 4; ++i) {
                int idx = i * 256 + tid;
                int row = idx >> 3, c8 = idx & 7;
                ta[i] = *(const v8bf*)&A [(size_t)(m0 + row) * K + k0 + 64 + c8 * 8];
                tb[i] = *(const v8bf*)&Bw[(size_t)(n0 + row) * K + k0 + 64 + c8 * 8];
            }
        }
#endif
        if (k0 + 128 < K) {  // warm L2 for the slice after next
            __builtin_prefetch(&A [(size_t)(m0 + (tid >> 1)) * K + k0 + 128], 0, 1);
            __builtin_prefetch(&Bw[(size_t)(n0 + (tid >> 1)) * K + k0 + 128], 0, 1);
        }

        // compute on current buffer
        #pragma unroll
        for (int kc = 0; kc < 2; ++kc) {
            v16bf af[2];
            #pragma unroll
            for (int mi = 0; mi < 2; ++mi) {
                int r = wrow * 32 + mi * 16 + l16;
                af[mi] = *(const v16bf*)&As[buf][r][kc * 32 + 16 * hl];
            }
            v16bf bfr[4];
            #pragma unroll
            for (int ni = 0; ni < 4; ++ni) {
                int c = wcol * 64 + ni * 16 + l16;
                bfr[ni] = *(const v16bf*)&Bs[buf][c][kc * 32 + 16 * hl];
            }
            #pragma unroll
            for (int mi = 0; mi < 2; ++mi)
                #pragma unroll
                for (int ni = 0; ni < 4; ++ni)
                    acc[mi][ni] = wmma_bf16(af[mi], bfr[ni], acc[mi][ni]);
        }

        if (haveNext) {
#if HAVE_TDM
            if (wave == 0) wait_tensorcnt0();
#else
            #pragma unroll
            for (int i = 0; i < 4; ++i) {
                int idx = i * 256 + tid;
                int row = idx >> 3, c8 = idx & 7;
                *(v8bf*)&As[buf ^ 1][row][c8 * 8] = ta[i];
                *(v8bf*)&Bs[buf ^ 1][row][c8 * 8] = tb[i];
            }
#endif
            __syncthreads();
        }
        buf ^= 1;
    }

    #pragma unroll
    for (int mi = 0; mi < 2; ++mi)
        #pragma unroll
        for (int ni = 0; ni < 4; ++ni)
            #pragma unroll
            for (int r = 0; r < 8; ++r) {
                int row = m0 + wrow * 32 + mi * 16 + r + 8 * hl;
                int col = n0 + wcol * 64 + ni * 16 + l16;
                C[(size_t)row * N + col] = acc[mi][ni][r];
            }
}

// ---------------------------------------------------------------- RMS-norm + RoPE
__launch_bounds__(128)
__global__ void qkv_post(const float* __restrict__ qkv,
                         const float* __restrict__ qg, const float* __restrict__ kg,
                         bf16_t* __restrict__ qb, bf16_t* __restrict__ kb,
                         bf16_t* __restrict__ vb) {
    const int idx = blockIdx.x;
    const int n  = idx % SEQ;
    const int hh = (idx / SEQ) % N_HEADS;
    const int b  = idx / (SEQ * N_HEADS);
    const int dd = threadIdx.x;

    const size_t rb = ((size_t)b * SEQ + n) * (3 * DMODEL);
    float qv = qkv[rb + 0 * DMODEL + hh * DHEAD + dd];
    float kv = qkv[rb + 1 * DMODEL + hh * DHEAD + dd];
    float vv = qkv[rb + 2 * DMODEL + hh * DHEAD + dd];

    __shared__ float red[2][4];
    float sq = qv * qv, sk = kv * kv;
    #pragma unroll
    for (int off = 16; off > 0; off >>= 1) {
        sq += __shfl_xor(sq, off);
        sk += __shfl_xor(sk, off);
    }
    const int lane = threadIdx.x & 31, w = threadIdx.x >> 5;
    if (lane == 0) { red[0][w] = sq; red[1][w] = sk; }
    __syncthreads();
    sq = red[0][0] + red[0][1] + red[0][2] + red[0][3];
    sk = red[1][0] + red[1][1] + red[1][2] + red[1][3];
    const float rq = rsqrtf(sq * (1.0f / DHEAD) + 1e-6f);
    const float rk = rsqrtf(sk * (1.0f / DHEAD) + 1e-6f);
    qv = qv * rq * qg[dd];
    kv = kv * rk * kg[dd];

    const int i = dd >> 1;
    const float freq = __powf(10000.0f, -(float)(2 * i) * (1.0f / DHEAD));
    float s, c;
    __sincosf((float)n * freq, &s, &c);
    const float qp = __shfl_xor(qv, 1);
    const float kp = __shfl_xor(kv, 1);
    float qr, kr;
    if ((dd & 1) == 0) { qr = qv * c - qp * s; kr = kv * c - kp * s; }
    else               { qr = qp * s + qv * c; kr = kp * s + kv * c; }

    const size_t ob = (((size_t)b * N_HEADS + hh) * SEQ + n) * DHEAD + dd;
    qb[ob] = (bf16_t)qr;
    kb[ob] = (bf16_t)kr;
    vb[ob] = (bf16_t)vv;
}

// ---------------------------------------------------------------- causal flash attention
__launch_bounds__(256)
__global__ void flash_attn(const bf16_t* __restrict__ Q, const bf16_t* __restrict__ Kg,
                           const bf16_t* __restrict__ V, bf16_t* __restrict__ ctx) {
    __shared__ __align__(32) bf16_t KP[128][DHEAD];  // K tile (TDM dest), reused as P scratch
    __shared__ __align__(32) bf16_t Vt[DHEAD][128];  // V tile transposed (d-major)

    const int qblk = blockIdx.x;
    const int h    = blockIdx.y;
    const int b    = blockIdx.z;
    const int tid  = threadIdx.x;
    const int lane = tid & 31;
    const int wave = tid >> 5;
    const int hl   = lane >> 4;
    const int l16  = lane & 15;
    const size_t headoff = (((size_t)b * N_HEADS + h) * SEQ) * DHEAD;
    const int q0 = qblk * 128 + wave * 16;
    const float scale = 0.0883883476483184f;  // 1/sqrt(128)

#if HAVE_TDM
    const unsigned kp_lds = lds_addr_of(&KP[0][0]);
#endif

    v16bf qa[4];
    {
        const bf16_t* qrow = Q + headoff + (size_t)(q0 + l16) * DHEAD;
        #pragma unroll
        for (int c = 0; c < 4; ++c) {
            qa[c] = *(const v16bf*)&qrow[c * 32 + 16 * hl];
        }
    }

    v8f o[8];
    float mrow[8], lrow[8];
    {
        v8f z = {};
        #pragma unroll
        for (int dn = 0; dn < 8; ++dn) o[dn] = z;
        #pragma unroll
        for (int r = 0; r < 8; ++r) { mrow[r] = -1e30f; lrow[r] = 0.0f; }
    }

    for (int j = 0; j <= qblk; ++j) {
        const int kv0 = j * 128;
        __syncthreads();

#if HAVE_TDM
        if (wave == 0) {
            tdm_load_2d_b16(kp_lds,
                            (unsigned long long)(size_t)(Kg + headoff + (size_t)kv0 * DHEAD),
                            DHEAD, SEQ, 128, 128, DHEAD);
        }
#else
        {
            const bf16_t* kb_ = Kg + headoff + (size_t)kv0 * DHEAD;
            #pragma unroll
            for (int i = 0; i < 8; ++i) {
                int idx = i * 256 + tid;
                int row = idx >> 3, c8 = idx & 7;
                *(v8bf*)&KP[row][c8 * 8] =
                    *(const v8bf*)&kb_[(size_t)row * DHEAD + c8 * 8];
            }
        }
#endif
        // stage V transposed (overlaps with the TDM transfer)
        {
            const bf16_t* vb_ = V + headoff + (size_t)kv0 * DHEAD;
            #pragma unroll
            for (int i = 0; i < 4; ++i) {
                int idx = i * 256 + tid;
                int rp = idx >> 4;
                int dc = idx & 15;
                v8bf a = *(const v8bf*)&vb_[(size_t)(2 * rp) * DHEAD + dc * 8];
                v8bf bb = *(const v8bf*)&vb_[(size_t)(2 * rp + 1) * DHEAD + dc * 8];
                #pragma unroll
                for (int e = 0; e < 8; ++e) {
                    union { bf16_t hv[2]; unsigned u; } pk;
                    pk.hv[0] = a[e]; pk.hv[1] = bb[e];
                    *(unsigned*)&Vt[dc * 8 + e][2 * rp] = pk.u;
                }
            }
        }
#if HAVE_TDM
        if (wave == 0) wait_tensorcnt0();
#endif
        __syncthreads();

        // S = Q * K^T
        v8f S[8];
        #pragma unroll
        for (int nt = 0; nt < 8; ++nt) {
            v8f s = {};
            #pragma unroll
            for (int c = 0; c < 4; ++c) {
                v16bf kb2 = *(const v16bf*)&KP[nt * 16 + l16][c * 32 + 16 * hl];
                s = wmma_bf16(qa[c], kb2, s);
            }
            S[nt] = s;
        }

        #pragma unroll
        for (int nt = 0; nt < 8; ++nt)
            #pragma unroll
            for (int r = 0; r < 8; ++r) {
                float v = S[nt][r] * scale;
                if (j == qblk && (kv0 + nt * 16 + l16) > (q0 + r + 8 * hl)) v = -3e38f;
                S[nt][r] = v;
            }

        float newm[8], corr[8];
        #pragma unroll
        for (int r = 0; r < 8; ++r) {
            float t = S[0][r];
            #pragma unroll
            for (int nt = 1; nt < 8; ++nt) t = fmaxf(t, S[nt][r]);
            #pragma unroll
            for (int off = 8; off > 0; off >>= 1) t = fmaxf(t, __shfl_xor(t, off));
            newm[r] = fmaxf(mrow[r], t);
            corr[r] = __expf(mrow[r] - newm[r]);
            mrow[r] = newm[r];
        }
        #pragma unroll
        for (int nt = 0; nt < 8; ++nt)
            #pragma unroll
            for (int r = 0; r < 8; ++r)
                S[nt][r] = __expf(S[nt][r] - newm[r]);
        #pragma unroll
        for (int r = 0; r < 8; ++r) {
            float t = 0.0f;
            #pragma unroll
            for (int nt = 0; nt < 8; ++nt) t += S[nt][r];
            #pragma unroll
            for (int off = 8; off > 0; off >>= 1) t += __shfl_xor(t, off);
            lrow[r] = lrow[r] * corr[r] + t;
        }
        #pragma unroll
        for (int dn = 0; dn < 8; ++dn)
            #pragma unroll
            for (int r = 0; r < 8; ++r) o[dn][r] *= corr[r];

        __syncthreads();  // all waves done reading K tile -> reuse as P scratch
        #pragma unroll
        for (int nt = 0; nt < 8; ++nt)
            #pragma unroll
            for (int r = 0; r < 8; ++r)
                KP[wave * 16 + r + 8 * hl][nt * 16 + l16] = (bf16_t)S[nt][r];

        #pragma unroll
        for (int c = 0; c < 4; ++c) {
            const int prow = wave * 16 + l16;
            v16bf pa = *(const v16bf*)&KP[prow][c * 32 + 16 * hl];
            #pragma unroll
            for (int dn = 0; dn < 8; ++dn) {
                v16bf vbf = *(const v16bf*)&Vt[dn * 16 + l16][c * 32 + 16 * hl];
                o[dn] = wmma_bf16(pa, vbf, o[dn]);
            }
        }
    }

    float inv[8];
    #pragma unroll
    for (int r = 0; r < 8; ++r) inv[r] = 1.0f / lrow[r];
    #pragma unroll
    for (int dn = 0; dn < 8; ++dn)
        #pragma unroll
        for (int r = 0; r < 8; ++r) {
            int row = q0 + r + 8 * hl;
            int col = h * DHEAD + dn * 16 + l16;
            ctx[((size_t)b * SEQ + row) * DMODEL + col] = (bf16_t)(o[dn][r] * inv[r]);
        }
}

// ---------------------------------------------------------------- launcher
extern "C" void kernel_launch(void* const* d_in, const int* in_sizes, int n_in,
                              void* d_out, int out_size, void* d_ws, size_t ws_size,
                              hipStream_t stream) {
    (void)in_sizes; (void)n_in; (void)out_size; (void)ws_size;
    const float* x     = (const float*)d_in[0];
    const float* w_qkv = (const float*)d_in[1];
    const float* w_out = (const float*)d_in[2];
    const float* qn_g  = (const float*)d_in[3];
    const float* kn_g  = (const float*)d_in[4];
    float* out = (float*)d_out;

    char* ws = (char*)d_ws;
    size_t off = 0;
    auto alloc = [&](size_t bytes) -> void* {
        void* p = ws + off;
        off += (bytes + 255) & ~(size_t)255;
        return p;
    };
    const size_t nTok = (size_t)BATCH * SEQ;
    bf16_t* xb    = (bf16_t*)alloc(nTok * DMODEL * 2);
    bf16_t* wqkvb = (bf16_t*)alloc((size_t)3 * DMODEL * DMODEL * 2);
    bf16_t* woutb = (bf16_t*)alloc((size_t)DMODEL * DMODEL * 2);
    float*  qkvf  = (float*) alloc(nTok * 3 * DMODEL * 4);
    bf16_t* qb    = (bf16_t*)alloc(nTok * DMODEL * 2);
    bf16_t* kb    = (bf16_t*)alloc(nTok * DMODEL * 2);
    bf16_t* vb    = (bf16_t*)alloc(nTok * DMODEL * 2);
    bf16_t* ctxb  = (bf16_t*)alloc(nTok * DMODEL * 2);

    cvt_f32_bf16<<<2048, 256, 0, stream>>>(x, xb, (int)(nTok * DMODEL));
    cvt_f32_bf16<<<2048, 256, 0, stream>>>(w_qkv, wqkvb, 3 * DMODEL * DMODEL);
    cvt_f32_bf16<<<2048, 256, 0, stream>>>(w_out, woutb, DMODEL * DMODEL);

    gemm_bf16_tn<<<dim3(3 * DMODEL / 128, (int)nTok / 128), 256, 0, stream>>>(
        xb, wqkvb, qkvf, (int)nTok, 3 * DMODEL, DMODEL);

    qkv_post<<<BATCH * N_HEADS * SEQ, 128, 0, stream>>>(qkvf, qn_g, kn_g, qb, kb, vb);

    flash_attn<<<dim3(SEQ / 128, N_HEADS, BATCH), 256, 0, stream>>>(qb, kb, vb, ctxb);

    gemm_bf16_tn<<<dim3(DMODEL / 128, (int)nTok / 128), 256, 0, stream>>>(
        ctxb, woutb, out, (int)nTok, DMODEL, DMODEL);
}
